// ParallelCoAttentionNetwork_26396869001457
// MI455X (gfx1250) — compile-verified, matrix operand order checked by reading, not run on hardware
//
#include <hip/hip_runtime.h>
#include <hip/hip_bf16.h>

typedef __bf16 bf16;
typedef __attribute__((ext_vector_type(16))) __bf16 v16bf;
typedef __attribute__((ext_vector_type(8)))  __bf16 v8bf;
typedef __attribute__((ext_vector_type(8)))  float   v8f;

union F16frag { v16bf v; v8bf h[2]; };
union V8acc   { v8f v; float f[8]; };
union F8out   { v8bf v; bf16 e[8]; uint4 u; };

#define AS_LD 48   // 32 cols + 16 pad, row stride 96B (16B aligned)
#define BS_LD 48

// ---------------------------------------------------------------- pack kernels
__global__ void f32_to_bf16_k(const float* __restrict__ s, bf16* __restrict__ d, size_t n) {
  size_t i = (size_t)blockIdx.x * blockDim.x + threadIdx.x;
  size_t stride = (size_t)gridDim.x * blockDim.x;
  for (; i < n; i += stride) d[i] = (bf16)s[i];
}

__global__ void zero_f32_k(float* __restrict__ p, int n) {
  int i = blockIdx.x * blockDim.x + threadIdx.x;
  if (i < n) p[i] = 0.f;
}

// V [B,H,R] f32 -> Vt [B,R,H] bf16   (32x32 LDS tile transpose)
__global__ void transpose_pack_v_k(const float* __restrict__ V, bf16* __restrict__ Vt) {
  __shared__ float t[32][33];
  const int r0 = blockIdx.x * 32, h0 = blockIdx.y * 32, b = blockIdx.z;
  const float* Vb = V + (size_t)b * 1024 * 576;
  bf16* Vtb = Vt + (size_t)b * 576 * 1024;
#pragma unroll
  for (int i = 0; i < 4; ++i) {
    int h = threadIdx.y + i * 8;
    t[h][threadIdx.x] = Vb[(size_t)(h0 + h) * 576 + r0 + threadIdx.x];
  }
  __syncthreads();
#pragma unroll
  for (int i = 0; i < 4; ++i) {
    int rr = threadIdx.y + i * 8;
    Vtb[(size_t)(r0 + rr) * 1024 + h0 + threadIdx.x] = (bf16)t[threadIdx.x][rr];
  }
}

// ---------------------------------------------------------------- WMMA mainloop
// Computes 128x64 tile at (m0,n0): A [M,K] row-major bf16, B given as B^T [N,K] row-major bf16.
// Each of 8 waves (4 in M x 2 in N) owns a 32x32 tile = 2x2 wmma accumulators.
__device__ __forceinline__ void wmma_mainloop(
    const bf16* __restrict__ Ab, int lda,
    const bf16* __restrict__ Bb, int ldb,
    int K, int m0, int n0,
    bf16* As, bf16* Bs, V8acc acc[2][2]) {
  const int tid = threadIdx.x;
  const int lane = tid & 31, wave = tid >> 5;
  const int wm = wave >> 1, wn = wave & 1;
  const int lm = lane & 15, half = lane >> 4;

  for (int k0 = 0; k0 < K; k0 += 32) {
    __syncthreads();
    // stage A tile 128x32 (512 chunks of 8 bf16; 2 per thread)
#pragma unroll
    for (int c = 0; c < 2; ++c) {
      int idx = tid + c * 256;
      int row = idx >> 2, col = (idx & 3) << 3;
      *reinterpret_cast<uint4*>(&As[row * AS_LD + col]) =
          *reinterpret_cast<const uint4*>(&Ab[(size_t)(m0 + row) * lda + k0 + col]);
    }
    // stage B tile (N-major) 64x32 (1 chunk per thread)
    {
      int row = tid >> 2, col = (tid & 3) << 3;
      *reinterpret_cast<uint4*>(&Bs[row * BS_LD + col]) =
          *reinterpret_cast<const uint4*>(&Bb[(size_t)(n0 + row) * ldb + k0 + col]);
    }
    __syncthreads();

    // gather fragments per documented 16-bit A layout:
    // lane halves: elems 0..7 -> K = half*8 + i ; elems 8..15 -> K = 16 + half*8 + i
    F16frag af[2], bfrag[2];
#pragma unroll
    for (int s = 0; s < 2; ++s) {
      int ar = (wm * 32 + s * 16 + lm) * AS_LD;
      af[s].h[0] = *reinterpret_cast<const v8bf*>(&As[ar + half * 8]);
      af[s].h[1] = *reinterpret_cast<const v8bf*>(&As[ar + 16 + half * 8]);
      int br = (wn * 32 + s * 16 + lm) * BS_LD;
      bfrag[s].h[0] = *reinterpret_cast<const v8bf*>(&Bs[br + half * 8]);
      bfrag[s].h[1] = *reinterpret_cast<const v8bf*>(&Bs[br + 16 + half * 8]);
    }
#pragma unroll
    for (int sm = 0; sm < 2; ++sm)
#pragma unroll
      for (int sn = 0; sn < 2; ++sn)
        acc[sm][sn].v = __builtin_amdgcn_wmma_f32_16x16x32_bf16(
            false, af[sm].v, false, bfrag[sn].v, (short)0, acc[sm][sn].v, false, false);
  }
  __syncthreads();
}

// ---------------------------------------------------------------- GEMM with stores
// MODE bit0: store normal [M,N] bf16 ; bit1: store transposed [N,M] bf16
template <int MODE>
__global__ __launch_bounds__(256) void gemm_store_k(
    const bf16* __restrict__ A, long long sA, int lda,
    const bf16* __restrict__ Bm, long long sB, int ldb,
    bf16* __restrict__ outN, long long sON, int ldoN,
    bf16* __restrict__ outT, long long sOT, int ldoT,
    int K) {
  __shared__ __align__(16) bf16 As[128 * AS_LD];
  __shared__ __align__(16) bf16 Bs[64 * BS_LD];
  const int tid = threadIdx.x, b = blockIdx.z;
  const int m0 = blockIdx.y * 128, n0 = blockIdx.x * 64;
  const int lane = tid & 31, wave = tid >> 5;
  const int wm = wave >> 1, wn = wave & 1;
  const int lm = lane & 15, half = lane >> 4;

  V8acc acc[2][2] = {};
  wmma_mainloop(A + (size_t)b * sA, lda, Bm + (size_t)b * sB, ldb, K, m0, n0, As, Bs, acc);

  if constexpr (MODE & 1) {
    bf16* op = outN + (size_t)b * sON;
#pragma unroll
    for (int sm = 0; sm < 2; ++sm) {
      int Mb = m0 + wm * 32 + sm * 16 + half * 8;
#pragma unroll
      for (int sn = 0; sn < 2; ++sn) {
        int Nc = n0 + wn * 32 + sn * 16 + lm;
#pragma unroll
        for (int j = 0; j < 8; ++j)
          op[(size_t)(Mb + j) * ldoN + Nc] = (bf16)acc[sm][sn].f[j];
      }
    }
  }
  if constexpr (MODE & 2) {
    bf16* op = outT + (size_t)b * sOT;
#pragma unroll
    for (int sm = 0; sm < 2; ++sm) {
      int Mb = m0 + wm * 32 + sm * 16 + half * 8;
#pragma unroll
      for (int sn = 0; sn < 2; ++sn) {
        int Nc = n0 + wn * 32 + sn * 16 + lm;
        F8out t;
#pragma unroll
        for (int j = 0; j < 8; ++j) t.e[j] = (bf16)acc[sm][sn].f[j];
        *reinterpret_cast<uint4*>(&op[(size_t)Nc * ldoT + Mb]) = t.u;
      }
    }
  }
}

// ---------------------------------------------------------------- fused GEMM + tanh + w·H
// s[b, n] += sum_c w[c] * tanh(Add[c,n] + (A@B)[c,n])   (never materializes H)
// After the WMMA mainloop the As/Bs LDS is dead; reuse it to stage the 128x64
// Add tile with coalesced b128 loads instead of strided scalar bf16 loads.
__global__ __launch_bounds__(256) void gemm_tanh_dot_k(
    const bf16* __restrict__ A, long long sA, int lda,
    const bf16* __restrict__ Bm, long long sB, int ldb,
    const bf16* __restrict__ Add, long long sAdd, int ldAdd,
    const float* __restrict__ wvec,
    float* __restrict__ sOut, int strideS,
    int K) {
  __shared__ __align__(16) bf16 smem[128 * AS_LD + 64 * BS_LD];  // 9216 bf16 = 18.4 KB
  bf16* As = smem;
  bf16* Bs = smem + 128 * AS_LD;
  __shared__ float wlds[128];
  __shared__ float red[4][64];
  const int tid = threadIdx.x, b = blockIdx.z;
  const int m0 = blockIdx.y * 128, n0 = blockIdx.x * 64;
  const int lane = tid & 31, wave = tid >> 5;
  const int wm = wave >> 1, wn = wave & 1;
  const int lm = lane & 15, half = lane >> 4;

  if (tid < 128) wlds[tid] = wvec[m0 + tid];

  V8acc acc[2][2] = {};
  wmma_mainloop(A + (size_t)b * sA, lda, Bm + (size_t)b * sB, ldb, K, m0, n0, As, Bs, acc);
  // mainloop ends with __syncthreads(): smem is reusable now

  // stage Add tile [128 rows][64 cols] bf16, row stride 72 (144B, 16B aligned)
  constexpr int ADD_LD = 72;  // 128*72 = 9216 elems == sizeof(smem)
  const bf16* addp = Add + (size_t)b * sAdd;
#pragma unroll
  for (int c = 0; c < 4; ++c) {
    int idx = tid + c * 256;               // 1024 chunks of 8 bf16
    int row = idx >> 3, col = (idx & 7) << 3;
    *reinterpret_cast<uint4*>(&smem[row * ADD_LD + col]) =
        *reinterpret_cast<const uint4*>(&addp[(size_t)(m0 + row) * ldAdd + n0 + col]);
  }
  __syncthreads();

  float part[2];
#pragma unroll
  for (int sn = 0; sn < 2; ++sn) {
    int nl = wn * 32 + sn * 16 + lm;  // local column within 64-tile
    float p = 0.f;
#pragma unroll
    for (int sm = 0; sm < 2; ++sm) {
      int Mloc = wm * 32 + sm * 16 + half * 8;
#pragma unroll
      for (int j = 0; j < 8; ++j) {
        float av = (float)smem[(Mloc + j) * ADD_LD + nl];
        p += wlds[Mloc + j] * tanhf(av + acc[sm][sn].f[j]);
      }
    }
    p += __shfl_down(p, 16);  // combine half-lanes (same column)
    part[sn] = p;
  }
  if (lane < 16) {
    red[wm][wn * 32 + lm]      = part[0];
    red[wm][wn * 32 + 16 + lm] = part[1];
  }
  __syncthreads();
  if (tid < 64) {
    float s = red[0][tid] + red[1][tid] + red[2][tid] + red[3][tid];
    atomicAdd(&sOut[(size_t)b * strideS + n0 + tid], s);
  }
}

// ---------------------------------------------------------------- softmaxes
__global__ void softmax_row_k(const float* __restrict__ s, float* __restrict__ out, int n) {
  __shared__ float red[256];
  const int b = blockIdx.x, tid = threadIdx.x;
  const float* row = s + (size_t)b * n;
  float m = -1e30f;
  for (int i = tid; i < n; i += 256) m = fmaxf(m, row[i]);
  red[tid] = m; __syncthreads();
  for (int o = 128; o > 0; o >>= 1) { if (tid < o) red[tid] = fmaxf(red[tid], red[tid + o]); __syncthreads(); }
  m = red[0]; __syncthreads();
  float sum = 0.f;
  for (int i = tid; i < n; i += 256) sum += expf(row[i] - m);
  red[tid] = sum; __syncthreads();
  for (int o = 128; o > 0; o >>= 1) { if (tid < o) red[tid] += red[tid + o]; __syncthreads(); }
  float inv = 1.f / red[0];
  for (int i = tid; i < n; i += 256) out[(size_t)b * n + i] = expf(row[i] - m) * inv;
}

// softmax(s) then masked softmax-of-softmax (faithful to reference)
__global__ void softmax_q_masked_k(const float* __restrict__ s, const int* __restrict__ lens,
                                   float* __restrict__ out) {
  const int L = 512;
  __shared__ float a[512];
  __shared__ float red[256];
  const int b = blockIdx.x, tid = threadIdx.x;
  const float* row = s + (size_t)b * L;
  float m = -1e30f;
  for (int i = tid; i < L; i += 256) m = fmaxf(m, row[i]);
  red[tid] = m; __syncthreads();
  for (int o = 128; o > 0; o >>= 1) { if (tid < o) red[tid] = fmaxf(red[tid], red[tid + o]); __syncthreads(); }
  m = red[0]; __syncthreads();
  float sum = 0.f;
  for (int i = tid; i < L; i += 256) sum += expf(row[i] - m);
  red[tid] = sum; __syncthreads();
  for (int o = 128; o > 0; o >>= 1) { if (tid < o) red[tid] += red[tid + o]; __syncthreads(); }
  float inv = 1.f / red[0]; __syncthreads();
  for (int i = tid; i < L; i += 256) a[i] = expf(row[i] - m) * inv;
  __syncthreads();
  const int len = lens[b];  // guaranteed >= 256
  float m2 = -1e30f;
  for (int i = tid; i < L; i += 256) if (i < len) m2 = fmaxf(m2, a[i]);
  red[tid] = m2; __syncthreads();
  for (int o = 128; o > 0; o >>= 1) { if (tid < o) red[tid] = fmaxf(red[tid], red[tid + o]); __syncthreads(); }
  m2 = red[0]; __syncthreads();
  float s2 = 0.f;
  for (int i = tid; i < L; i += 256) if (i < len) s2 += expf(a[i] - m2);
  red[tid] = s2; __syncthreads();
  for (int o = 128; o > 0; o >>= 1) { if (tid < o) red[tid] += red[tid + o]; __syncthreads(); }
  float inv2 = 1.f / red[0];
  for (int i = tid; i < L; i += 256)
    out[(size_t)b * L + i] = (i < len) ? expf(a[i] - m2) * inv2 : 0.f;
}

// ---------------------------------------------------------------- final GEMVs (fp32)
// v[b,h] = sum_r a_v[b,r] * V[b,h,r]   (one wave per row, lanes stride R)
__global__ void gemv_v_k(const float* __restrict__ V, const float* __restrict__ av,
                         float* __restrict__ vout) {
  const int gw = blockIdx.x * 8 + (threadIdx.x >> 5);
  const int b = gw >> 10, h = gw & 1023;
  const int lane = threadIdx.x & 31;
  const float* row = V + ((size_t)b * 1024 + h) * 576;
  const float* a = av + (size_t)b * 576;
  float acc = 0.f;
  for (int r = lane; r < 576; r += 32) acc += row[r] * a[r];
  for (int o = 16; o > 0; o >>= 1) acc += __shfl_down(acc, o);
  if (lane == 0) vout[gw] = acc;
}

// q[b,h] = sum_l ma[b,l] * Q[b,l,h]   (threads over h -> coalesced)
__global__ void gemv_q_k(const float* __restrict__ Q, const float* __restrict__ ma,
                         float* __restrict__ qout) {
  __shared__ float ml[512];
  const int b = blockIdx.y;
  const int h = blockIdx.x * 256 + threadIdx.x;
  for (int i = threadIdx.x; i < 512; i += 256) ml[i] = ma[(size_t)b * 512 + i];
  __syncthreads();
  const float* Qb = Q + (size_t)b * 512 * 1024;
  float acc = 0.f;
  for (int l = 0; l < 512; ++l) acc += ml[l] * Qb[(size_t)l * 1024 + h];
  qout[(size_t)b * 1024 + h] = acc;
}

// ---------------------------------------------------------------- host launcher
extern "C" void kernel_launch(void* const* d_in, const int* in_sizes, int n_in,
                              void* d_out, int out_size, void* d_ws, size_t ws_size,
                              hipStream_t stream) {
  constexpr int Bn = 64, Hd = 1024, CDd = 1024, Rd = 576, Ln = 512;
  const float* V    = (const float*)d_in[0];
  const float* Q    = (const float*)d_in[1];
  const int*   Qlen = (const int*)d_in[2];
  const float* Wb   = (const float*)d_in[3];
  const float* Wv   = (const float*)d_in[4];
  const float* Wq   = (const float*)d_in[5];
  const float* whv  = (const float*)d_in[6];
  const float* whq  = (const float*)d_in[7];
  float* out = (float*)d_out;

  char* ws = (char*)d_ws;
  size_t off = 0;
  auto alloc = [&](size_t bytes) -> void* {
    off = (off + 255) & ~(size_t)255;
    void* p = ws + off;
    off += bytes;
    return p;
  };
  bf16* wb_bf = (bf16*)alloc((size_t)Hd * Hd * 2);
  bf16* wv_bf = (bf16*)alloc((size_t)Hd * Hd * 2);
  bf16* wq_bf = (bf16*)alloc((size_t)Hd * Hd * 2);
  bf16* q_bf  = (bf16*)alloc((size_t)Bn * Ln * Hd * 2);
  bf16* vt    = (bf16*)alloc((size_t)Bn * Rd * Hd * 2);  // later aliased by C + Ct
  bf16* wbvt  = (bf16*)alloc((size_t)Bn * Rd * Hd * 2);
  bf16* wvv   = (bf16*)alloc((size_t)Bn * CDd * Rd * 2);
  bf16* wqq   = (bf16*)alloc((size_t)Bn * CDd * Ln * 2);
  float* s_v  = (float*)alloc((size_t)Bn * Rd * 4);
  float* s_q  = (float*)alloc((size_t)Bn * Ln * 4);  // contiguous after s_v
  bf16* Cb = vt;                                     // [B,L,R]  (Vt is dead by then)
  bf16* Ct = vt + (size_t)Bn * Ln * Rd;              // [B,R,L]

  // --- pack to bf16 / transpose / zero accumulators
  f32_to_bf16_k<<<1024, 256, 0, stream>>>(Wb, wb_bf, (size_t)Hd * Hd);
  f32_to_bf16_k<<<1024, 256, 0, stream>>>(Wv, wv_bf, (size_t)Hd * Hd);
  f32_to_bf16_k<<<1024, 256, 0, stream>>>(Wq, wq_bf, (size_t)Hd * Hd);
  f32_to_bf16_k<<<4096, 256, 0, stream>>>(Q, q_bf, (size_t)Bn * Ln * Hd);
  transpose_pack_v_k<<<dim3(Rd / 32, Hd / 32, Bn), dim3(32, 8), 0, stream>>>(V, vt);
  zero_f32_k<<<(Bn * (Rd + Ln) + 255) / 256, 256, 0, stream>>>(s_v, Bn * (Rd + Ln));

  // --- GEMM 1: WbV^T[b] = (W_b @ V_b)^T   [R,H] bf16
  gemm_store_k<2><<<dim3(Rd / 64, Hd / 128, Bn), 256, 0, stream>>>(
      wb_bf, 0, Hd, vt, (long long)Rd * Hd, Hd,
      nullptr, 0, 0, wbvt, (long long)Rd * Hd, Hd, Hd);
  // --- GEMM 2: WvV[b] = W_v @ V_b   [CD,R] bf16
  gemm_store_k<1><<<dim3(Rd / 64, CDd / 128, Bn), 256, 0, stream>>>(
      wv_bf, 0, Hd, vt, (long long)Rd * Hd, Hd,
      wvv, (long long)CDd * Rd, Rd, nullptr, 0, 0, Hd);
  // --- GEMM 3: WqQ[b] = W_q @ Q_b^T   [CD,L] bf16  (B^T = Q_b itself)
  gemm_store_k<1><<<dim3(Ln / 64, CDd / 128, Bn), 256, 0, stream>>>(
      wq_bf, 0, Hd, q_bf, (long long)Ln * Hd, Hd,
      wqq, (long long)CDd * Ln, Ln, nullptr, 0, 0, Hd);
  // --- GEMM 4: C[b] = Q_b @ WbV  -> store both C [L,R] and C^T [R,L]
  gemm_store_k<3><<<dim3(Rd / 64, Ln / 128, Bn), 256, 0, stream>>>(
      q_bf, (long long)Ln * Hd, Hd, wbvt, (long long)Rd * Hd, Hd,
      Cb, (long long)Ln * Rd, Rd, Ct, (long long)Rd * Ln, Ln, Hd);
  // --- fused 5: s_q[b,l] = sum_c w_hq[c]*tanh(WqQ[c,l] + sum_r WvV[c,r]*C[l,r])
  gemm_tanh_dot_k<<<dim3(Ln / 64, CDd / 128, Bn), 256, 0, stream>>>(
      wvv, (long long)CDd * Rd, Rd, Cb, (long long)Ln * Rd, Rd,
      wqq, (long long)CDd * Ln, Ln, whq, s_q, Ln, Rd);
  // --- fused 6: s_v[b,r] = sum_c w_hv[c]*tanh(WvV[c,r] + sum_l WqQ[c,l]*C[l,r])
  gemm_tanh_dot_k<<<dim3(Rd / 64, CDd / 128, Bn), 256, 0, stream>>>(
      wqq, (long long)CDd * Ln, Ln, Ct, (long long)Rd * Ln, Ln,
      wvv, (long long)CDd * Rd, Rd, whv, s_v, Rd, Ln);

  // --- softmaxes -> first two outputs
  float* a_v_out = out;                       // [B,1,R]
  float* maq_out = out + (size_t)Bn * Rd;     // [B,1,L]
  float* v_out   = maq_out + (size_t)Bn * Ln; // [B,H]
  float* q_out   = v_out + (size_t)Bn * Hd;   // [B,H]
  softmax_row_k<<<Bn, 256, 0, stream>>>(s_v, a_v_out, Rd);
  softmax_q_masked_k<<<Bn, 256, 0, stream>>>(s_q, Qlen, maq_out);

  // --- final weighted sums
  gemv_v_k<<<Bn * Hd / 8, 256, 0, stream>>>(V, a_v_out, v_out);
  gemv_q_k<<<dim3(Hd / 256, Bn), 256, 0, stream>>>(Q, maq_out, q_out);
}
